// AttentionBlock_90460601189030
// MI455X (gfx1250) — matrix operand order repaired and checked by reading, not compiled, not run on hardware
//
#include <hip/hip_runtime.h>
#include <hip/hip_bf16.h>
#include <math.h>

// ---------------------------------------------------------------------------
// AttentionBlock for MI455X (gfx1250, wave32, WMMA 16x16x32 f16 -> f32 acc)
//   B=8, C=512, T=1024 (=32x32), 8 heads, ch=64, 32 groups
//   GEMM tiles staged to LDS via Tensor Data Mover (double-buffered),
//   synchronized with s_wait_tensorcnt + workgroup barriers.
// ---------------------------------------------------------------------------

typedef _Float16 half_t;
typedef __attribute__((ext_vector_type(16))) _Float16 v16h;
typedef __attribute__((ext_vector_type(8)))  _Float16 v8h;
typedef __attribute__((ext_vector_type(8)))  float    v8f;
typedef unsigned int u32x4 __attribute__((ext_vector_type(4)));
typedef int          i32x4 __attribute__((ext_vector_type(4)));
typedef int          i32x8 __attribute__((ext_vector_type(8)));

#define BB   8
#define CC   512
#define TT   1024
#define NH   8
#define CH   64
#define NG   32
#define QKV_SCALE 0.3535533905932738f   // 1/sqrt(sqrt(64))
#define INV_SQRT2 0.7071067811865476f

#if defined(__has_builtin)
#if __has_builtin(__builtin_amdgcn_tensor_load_to_lds)
#define AB_USE_TDM 1
#endif
#endif
#ifndef AB_USE_TDM
#define AB_USE_TDM 0
#endif

// ---- WMMA wrapper ----------------------------------------------------------
__device__ __forceinline__ v8f wmma_f16(v16h a, v16h b, v8f c) {
  return __builtin_amdgcn_wmma_f32_16x16x32_f16(
      /*neg_a=*/false, a, /*neg_b=*/false, b,
      /*c_mod=*/(short)0, c, /*reuse_a=*/false, /*reuse_b=*/false);
}

// ---- fragment loaders (contiguous k = 16*hi + s, same order in A and B) ------------------
// A matrix 16x32 f16, row-major source with leading dim ld (halfs).
// lane m (0..15) holds row m, K in {hi*16 .. hi*16+15}
__device__ __forceinline__ v16h load_frag_a(const half_t* p, int ld) {
  int lane = threadIdx.x & 31;
  int m = lane & 15, hi = lane >> 4;
  const half_t* row = p + (size_t)m * ld;
  return *(const v16h*)(row + hi * 16);
}

// B matrix 32x16 f16, column-major source: element (k,n) at p + n*ld + k.
__device__ __forceinline__ v16h load_frag_b(const half_t* p, int ld) {
  int lane = threadIdx.x & 31;
  int n = lane & 15, hi = lane >> 4;
  return *(const v16h*)(p + (size_t)n * ld + hi * 16);
}

#if AB_USE_TDM
// ---- Tensor Data Mover: 2D f16 tile (tile_w contiguous elems x tile_h rows)
// from global (row stride `stride` elems) into LDS at byte offset lds_off.
// D# layout per CDNA5 ISA ch.8: group0 {count, lds_addr, global_addr, type=2},
// group1 {data_size=2B, tensor dims, tile dims, dim0 stride}; groups 2/3 = 0.
__device__ __forceinline__ void tdm_load_tile_f16(
    unsigned lds_off, const half_t* gptr, int tile_w, int tile_h, int stride) {
  unsigned long long ga = (unsigned long long)(size_t)gptr;
  u32x4 g0;
  g0[0] = 1u;                                    // count=1, user mode
  g0[1] = lds_off;                               // lds_addr (bytes)
  g0[2] = (unsigned)(ga & 0xFFFFFFFFu);          // global_addr[31:0]
  g0[3] = (unsigned)((ga >> 32) & 0x01FFFFFFu)   // global_addr[56:32]
        | 0x80000000u;                           // type=2 ("image")
  const unsigned long long dim0 = 1u << 30;      // interior tiles: no OOB
  const unsigned long long dim1 = 1u << 30;
  unsigned long long q0 = (1ull << 16)                          // data_size=1 (2B)
                        | ((dim0 & 0xFFFFull) << 48);           // tensor_dim0 lo
  unsigned long long q1 = ((dim0 >> 16) & 0xFFFFull)            // tensor_dim0 hi
                        | ((dim1 & 0xFFFFFFFFull) << 16)        // tensor_dim1
                        | (((unsigned long long)tile_w & 0xFFFFull) << 48); // tile_dim0
  unsigned long long q2 = ((unsigned long long)tile_h & 0xFFFFull)          // tile_dim1
                        | (((unsigned long long)(unsigned)stride & 0xFFFFFFFFull) << 32);
  unsigned long long q3 = 0ull;                  // stride hi16 = 0, dim1_stride = 0 (2D)
  i32x8 g1;
  g1[0] = (int)q0; g1[1] = (int)(q0 >> 32);
  g1[2] = (int)q1; g1[3] = (int)(q1 >> 32);
  g1[4] = (int)q2; g1[5] = (int)(q2 >> 32);
  g1[6] = (int)q3; g1[7] = (int)(q3 >> 32);
  i32x4 gz = {0, 0, 0, 0};
#if __clang_major__ >= 23
  i32x8 gz8 = {0, 0, 0, 0, 0, 0, 0, 0};
  __builtin_amdgcn_tensor_load_to_lds(g0, g1, gz, gz, gz8, 0);
#else
  __builtin_amdgcn_tensor_load_to_lds(g0, g1, gz, gz, 0);
#endif
}
#endif  // AB_USE_TDM

// ---------------------------------------------------------------------------
// K0: convert weights fp32 -> f16 (A-layout row-major, as given)
// ---------------------------------------------------------------------------
__global__ __launch_bounds__(256) void ab_convert_w(
    const float* __restrict__ qkv_w, const float* __restrict__ proj_w,
    half_t* __restrict__ qw16, half_t* __restrict__ pw16) {
  int i = blockIdx.x * 256 + threadIdx.x;
  if (i < 3 * CC * CC) qw16[i] = (half_t)qkv_w[i];
  if (i < CC * CC)     pw16[i] = (half_t)proj_w[i];
}

// ---------------------------------------------------------------------------
// K1: GroupNorm; writes xn16 in (b, t, c) order == B-layout for QKV GEMM
// ---------------------------------------------------------------------------
__global__ __launch_bounds__(256) void ab_groupnorm(
    const float* __restrict__ x, const float* __restrict__ gw,
    const float* __restrict__ gb, half_t* __restrict__ xn16) {
  int b = blockIdx.x >> 5;
  int g = blockIdx.x & 31;
  const float* xp = x + ((size_t)b * CC + g * 16) * TT;
  int tid = threadIdx.x;
  float s = 0.f, ss = 0.f;
  for (int i = tid; i < 16 * TT; i += 256) {
    float v = xp[i]; s += v; ss += v * v;
  }
  __shared__ float rs[256], rq[256];
  rs[tid] = s; rq[tid] = ss;
  __syncthreads();
  for (int off = 128; off; off >>= 1) {
    if (tid < off) { rs[tid] += rs[tid + off]; rq[tid] += rq[tid + off]; }
    __syncthreads();
  }
  const float inv_n = 1.0f / (16.0f * TT);
  float mean = rs[0] * inv_n;
  float var  = rq[0] * inv_n - mean * mean;
  float inv  = rsqrtf(var + 1e-6f);
  for (int i = tid; i < 16 * TT; i += 256) {
    int cl = i >> 10, t = i & (TT - 1);
    int c = g * 16 + cl;
    float v = (xp[i] - mean) * inv * gw[c] + gb[c];
    xn16[((size_t)b * TT + t) * CC + c] = (half_t)v;
  }
}

// ---------------------------------------------------------------------------
// GEMM core: 128x128 block tile, 8 waves (2x4), wave = 64x32 (4x2 WMMA tiles).
// TDM double-buffered LDS staging (wave 0 drives the DMA engine).
// ---------------------------------------------------------------------------
#define GEMM_TILE_BODY(Ag, Bg)                                                 \
  v8f acc[4][2];                                                               \
  _Pragma("unroll") for (int i = 0; i < 4; ++i)                                \
  _Pragma("unroll") for (int j = 0; j < 2; ++j) acc[i][j] = (v8f){};           \
  int buf = 0;                                                                 \
  TDM_PROLOGUE(Ag, Bg);                                                        \
  for (int k0 = 0; k0 < CC; k0 += 32) {                                        \
    TDM_STAGE(Ag, Bg);                                                         \
    v16h af[4], bf[2];                                                         \
    _Pragma("unroll") for (int i = 0; i < 4; ++i)                              \
      af[i] = load_frag_a(&As[buf][(wm + i * 16) * 32], 32);                   \
    _Pragma("unroll") for (int j = 0; j < 2; ++j)                              \
      bf[j] = load_frag_b(&Bs[buf][(wn + j * 16) * 32], 32);                   \
    _Pragma("unroll") for (int i = 0; i < 4; ++i)                              \
    _Pragma("unroll") for (int j = 0; j < 2; ++j)                              \
      acc[i][j] = wmma_f16(af[i], bf[j], acc[i][j]);                           \
    __syncthreads();                                                           \
    buf ^= 1;                                                                  \
  }

#if AB_USE_TDM
#define TDM_PROLOGUE(Ag, Bg)                                                   \
  if (w == 0) {                                                                \
    tdm_load_tile_f16((unsigned)(size_t)&As[0][0], (Ag), 32, 128, CC);         \
    tdm_load_tile_f16((unsigned)(size_t)&Bs[0][0], (Bg), 32, 128, CC);         \
  }
#define TDM_STAGE(Ag, Bg)                                                      \
  if (w == 0) {                                                                \
    if (k0 + 32 < CC) {                                                        \
      tdm_load_tile_f16((unsigned)(size_t)&As[buf ^ 1][0], (Ag) + k0 + 32, 32, \
                        128, CC);                                              \
      tdm_load_tile_f16((unsigned)(size_t)&Bs[buf ^ 1][0], (Bg) + k0 + 32, 32, \
                        128, CC);                                              \
      __builtin_amdgcn_s_wait_tensorcnt((short)2);                             \
    } else {                                                                   \
      __builtin_amdgcn_s_wait_tensorcnt((short)0);                             \
    }                                                                          \
  }                                                                            \
  __syncthreads();
#else
#define TDM_PROLOGUE(Ag, Bg)
#define TDM_STAGE(Ag, Bg)                                                      \
  *(v16h*)&As[buf][lr * 32 + lo] = *(const v16h*)((Ag) + (size_t)lr * CC + k0 + lo); \
  *(v16h*)&Bs[buf][lr * 32 + lo] = *(const v16h*)((Bg) + (size_t)lr * CC + k0 + lo); \
  __syncthreads();
#endif

// ---------------------------------------------------------------------------
// K2: QKV GEMM  (per batch: [1536 x 512] x [512 x 1024]); routes q/k/v.
// ---------------------------------------------------------------------------
__global__ __launch_bounds__(256) void ab_qkv_gemm(
    const half_t* __restrict__ qw16, const half_t* __restrict__ xn16,
    const float* __restrict__ qkv_b, half_t* __restrict__ q16,
    half_t* __restrict__ k16, half_t* __restrict__ v16g) {
  __shared__ half_t As[2][128 * 32];
  __shared__ half_t Bs[2][128 * 32];
  int blk = blockIdx.x;
  int nt = blk & 7;
  int mt = (blk >> 3) % 12;
  int b  = blk / 96;
  int m0 = mt * 128, n0 = nt * 128;
  int tid = threadIdx.x, w = tid >> 5, lane = tid & 31;
  int wm = (w & 1) * 64, wn = (w >> 1) * 32;
  int hi = lane >> 4, ln = lane & 15;
#if !AB_USE_TDM
  int lr = tid >> 1;
  int lo = (tid & 1) * 16;
#endif
  const half_t* Ag = qw16 + (size_t)m0 * CC;
  const half_t* Bg = xn16 + ((size_t)b * TT + n0) * CC;

  GEMM_TILE_BODY(Ag, Bg)

  // epilogue: bias + route into q/k/v head-major layouts
#pragma unroll
  for (int i = 0; i < 4; ++i)
#pragma unroll
    for (int j = 0; j < 2; ++j)
#pragma unroll
      for (int r = 0; r < 8; ++r) {
        int m = m0 + wm + i * 16 + r + 8 * hi;   // output channel o
        int n = n0 + wn + j * 16 + ln;           // t
        float val = acc[i][j][r] + qkv_b[m];
        int head = m / 192, rr = m % 192;
        size_t bh = (size_t)b * NH + head;
        if (rr < 64)
          q16[(bh * TT + n) * CH + rr] = (half_t)(val * QKV_SCALE);        // (bh,t,c)
        else if (rr < 128)
          k16[(bh * TT + n) * CH + (rr - 64)] = (half_t)(val * QKV_SCALE); // (bh,s,c)
        else
          v16g[(bh * CH + (rr - 128)) * TT + n] = (half_t)val;             // (bh,c,s)
      }
}

// ---------------------------------------------------------------------------
// K3: attention per head, 32-query-row tile; scores live entirely in LDS.
// ---------------------------------------------------------------------------
__global__ __launch_bounds__(256) void ab_attention(
    const half_t* __restrict__ q16, const half_t* __restrict__ k16,
    const half_t* __restrict__ v16g, half_t* __restrict__ h16) {
  extern __shared__ char smem[];
  float*  S   = (float*)smem;                               // [32][1024]
  half_t* W   = (half_t*)(smem + 32 * TT * 4);              // [32][1024]
  float*  red = (float*)(smem + 32 * TT * 4 + 32 * TT * 2); // [32][8]

  int bh = blockIdx.x >> 5;
  int tt = blockIdx.x & 31;
  int t0 = tt * 32;
  int tid = threadIdx.x, w = tid >> 5, lane = tid & 31;
  int hi = lane >> 4, ln = lane & 15;
  int mi = w & 1;      // which 16-row query subtile
  int nc = w >> 1;     // 256-wide key chunk

  // ---- stage 1: scores S = (q s)^T (k s), K=64 ----------------------------
  const half_t* qp = q16 + ((size_t)bh * TT + t0 + mi * 16) * CH;
  v16h aq0 = load_frag_a(qp, CH);
  v16h aq1 = load_frag_a(qp + 32, CH);
  for (int s0 = nc * 256; s0 < nc * 256 + 256; s0 += 16) {
    const half_t* kp = k16 + ((size_t)bh * TT + s0) * CH;
    v16h b0 = load_frag_b(kp, CH);
    v16h b1 = load_frag_b(kp + 32, CH);
    v8f sa = (v8f){};
    sa = wmma_f16(aq0, b0, sa);
    sa = wmma_f16(aq1, b1, sa);
#pragma unroll
    for (int r = 0; r < 8; ++r)
      S[(mi * 16 + r + 8 * hi) * TT + s0 + ln] = sa[r];
  }
  __syncthreads();

  // ---- stage 2: softmax (8 threads / row, 128 cols each) ------------------
  int row = tid >> 3, part = tid & 7;
  float* Sr = S + (size_t)row * TT + part * 128;
  float mx = -1e30f;
  for (int i = 0; i < 128; ++i) mx = fmaxf(mx, Sr[i]);
  red[row * 8 + part] = mx;
  __syncthreads();
  float rmx = red[row * 8];
#pragma unroll
  for (int i = 1; i < 8; ++i) rmx = fmaxf(rmx, red[row * 8 + i]);
  __syncthreads();
  float sum = 0.f;
  for (int i = 0; i < 128; ++i) {
    float e = __expf(Sr[i] - rmx);
    Sr[i] = e;
    sum += e;
  }
  red[row * 8 + part] = sum;
  __syncthreads();
  float rsum = 0.f;
#pragma unroll
  for (int i = 0; i < 8; ++i) rsum += red[row * 8 + i];
  float inv = 1.0f / rsum;
  half_t* Wr = W + (size_t)row * TT + part * 128;
  for (int i = 0; i < 128; ++i) Wr[i] = (half_t)(Sr[i] * inv);
  __syncthreads();

  // ---- stage 3: out = W x V^T, K=1024 -------------------------------------
  int ci = w >> 1;
  v8f oacc = (v8f){};
  const half_t* vcol = v16g + ((size_t)bh * CH + ci * 16) * TT;
  for (int k0 = 0; k0 < TT; k0 += 32) {
    v16h aw = load_frag_a(W + (size_t)(mi * 16) * TT + k0, TT);
    v16h bv = load_frag_b(vcol + k0, TT);
    oacc = wmma_f16(aw, bv, oacc);
  }
  int b = bh >> 3, hl = bh & 7;
#pragma unroll
  for (int r = 0; r < 8; ++r) {
    int t = t0 + mi * 16 + r + 8 * hi;
    int c = hl * CH + ci * 16 + ln;
    h16[((size_t)b * TT + t) * CC + c] = (half_t)oacc[r];  // (b, t, c)
  }
}

// ---------------------------------------------------------------------------
// K4: proj GEMM + bias + residual + 1/sqrt(2)
// ---------------------------------------------------------------------------
__global__ __launch_bounds__(256) void ab_proj_gemm(
    const half_t* __restrict__ pw16, const half_t* __restrict__ h16,
    const float* __restrict__ proj_b, const float* __restrict__ x,
    float* __restrict__ out) {
  __shared__ half_t As[2][128 * 32];
  __shared__ half_t Bs[2][128 * 32];
  int blk = blockIdx.x;
  int nt = blk & 7;
  int mt = (blk >> 3) & 3;
  int b  = blk >> 5;
  int m0 = mt * 128, n0 = nt * 128;
  int tid = threadIdx.x, w = tid >> 5, lane = tid & 31;
  int wm = (w & 1) * 64, wn = (w >> 1) * 32;
  int hi = lane >> 4, ln = lane & 15;
#if !AB_USE_TDM
  int lr = tid >> 1;
  int lo = (tid & 1) * 16;
#endif
  const half_t* Ag = pw16 + (size_t)m0 * CC;
  const half_t* Bg = h16 + ((size_t)b * TT + n0) * CC;

  GEMM_TILE_BODY(Ag, Bg)

#pragma unroll
  for (int i = 0; i < 4; ++i)
#pragma unroll
    for (int j = 0; j < 2; ++j)
#pragma unroll
      for (int r = 0; r < 8; ++r) {
        int m = m0 + wm + i * 16 + r + 8 * hi;   // o
        int n = n0 + wn + j * 16 + ln;           // t
        size_t oi = ((size_t)b * CC + m) * TT + n;
        out[oi] = (x[oi] + acc[i][j][r] + proj_b[m]) * INV_SQRT2;
      }
}

// ---------------------------------------------------------------------------
// launch
// ---------------------------------------------------------------------------
extern "C" void kernel_launch(void* const* d_in, const int* in_sizes, int n_in,
                              void* d_out, int out_size, void* d_ws, size_t ws_size,
                              hipStream_t stream) {
  const float* x      = (const float*)d_in[0];
  const float* gn_w   = (const float*)d_in[1];
  const float* gn_b   = (const float*)d_in[2];
  const float* qkv_w  = (const float*)d_in[3];
  const float* qkv_b  = (const float*)d_in[4];
  const float* proj_w = (const float*)d_in[5];
  const float* proj_b = (const float*)d_in[6];
  float* out = (float*)d_out;

  char* ws = (char*)d_ws;
  const size_t SZ_BTC = (size_t)BB * TT * CC * sizeof(half_t);   // 8 MB
  half_t* xn16 = (half_t*)(ws);
  half_t* q16  = (half_t*)(ws + SZ_BTC);
  half_t* k16  = (half_t*)(ws + 2 * SZ_BTC);
  half_t* v16g = (half_t*)(ws + 3 * SZ_BTC);
  half_t* h16  = (half_t*)(ws + 4 * SZ_BTC);
  half_t* qw16 = (half_t*)(ws + 5 * SZ_BTC);
  half_t* pw16 = (half_t*)(ws + 5 * SZ_BTC + (size_t)3 * CC * CC * sizeof(half_t));

  ab_convert_w<<<(3 * CC * CC + 255) / 256, 256, 0, stream>>>(qkv_w, proj_w, qw16, pw16);
  ab_groupnorm<<<BB * NG, 256, 0, stream>>>(x, gn_w, gn_b, xn16);
  ab_qkv_gemm<<<BB * 12 * 8, 256, 0, stream>>>(qw16, xn16, qkv_b, q16, k16, v16g);
  size_t smem = (size_t)32 * TT * 4 + (size_t)32 * TT * 2 + 32 * 8 * 4;
  ab_attention<<<BB * NH * (TT / 32), 256, smem, stream>>>(q16, k16, v16g, h16);
  ab_proj_gemm<<<BB * 4 * 8, 256, 0, stream>>>(pw16, h16, proj_b, x, out);
}
